// MultiHeadAttention_19052474925457
// MI455X (gfx1250) — matrix operand order repaired and checked by reading, not compiled, not run on hardware
//
#include <hip/hip_runtime.h>
#include <stdint.h>

// ---------------------------------------------------------------------------
// CDNA5 (gfx1250) relative-position MHA, bf16 WMMA (v_wmma_f32_16x16x32_bf16)
// Async global->LDS staging (ASYNCcnt) with double-buffered tiles so the DMA
// for K-step i+1 overlaps the WMMAs of step i; one barrier per step.
// ---------------------------------------------------------------------------

typedef __attribute__((ext_vector_type(16))) __bf16 v16bf;
typedef __attribute__((ext_vector_type(8)))  __bf16 v8bf;
typedef __attribute__((ext_vector_type(8)))  float  v8f;
typedef int vs4i __attribute__((vector_size(16)));

#define Bb 4
#define Ss 1024
#define Dd 768
#define Hh 12
#define DK 64
#define R2 129
#define R2P 160           // padded rel rows (5 x 32 K-steps)
#define NEGC 1.0e12f
#define L2E 1.44269504088896340736f

#if __has_builtin(__builtin_amdgcn_global_load_async_to_lds_b128) && \
    __has_builtin(__builtin_amdgcn_s_wait_asynccnt)
#define USE_ASYNC_LDS 1
#endif

// ---- staging helpers ------------------------------------------------------

static __device__ __forceinline__ void stage16(const __bf16* g, __bf16* l) {
#ifdef USE_ASYNC_LDS
  __builtin_amdgcn_global_load_async_to_lds_b128(
      (__attribute__((address_space(1))) vs4i*)(uintptr_t)g,
      (__attribute__((address_space(3))) vs4i*)(uint32_t)(uintptr_t)l,
      0, 0);
#else
  *(float4*)l = *(const float4*)g;
#endif
}
template <int N>
static __device__ __forceinline__ void stage_wait() {
#ifdef USE_ASYNC_LDS
  __builtin_amdgcn_s_wait_asynccnt(N);
#endif
}

// stage a 64x32 bf16 tile (LDS stride 40) with 128 threads
static __device__ __forceinline__ void stage_tile32(const __bf16* g, int gld,
                                                    __bf16* l, int tid) {
#pragma unroll
  for (int it = 0; it < 2; ++it) {
    int seg = tid + it * 128;
    int r = seg >> 2, c = (seg & 3) << 3;
    stage16(&g[r * gld + c], &l[r * 40 + c]);
  }
}
// stage a 64x64 bf16 tile (LDS stride 64) with 64 threads
static __device__ __forceinline__ void stage_tile64(const __bf16* g, int gld,
                                                    __bf16* l, int tid) {
#pragma unroll
  for (int it = 0; it < 8; ++it) {
    int seg = tid + it * 64;
    int r = seg >> 3, cc = (seg & 7) << 3;
    stage16(&g[r * gld + cc], &l[r * 64 + cc]);
  }
}

// ---- WMMA helpers ---------------------------------------------------------

static __device__ __forceinline__ v8f wmma_bf16(v16bf a, v16bf b, v8f c) {
  return __builtin_amdgcn_wmma_f32_16x16x32_bf16(false, a, false, b, (short)0, c,
                                                 false, false);
}

static __device__ __forceinline__ v16bf load2x8(const __bf16* p0, const __bf16* p1) {
  union { v16bf v; v8bf h[2]; } u;
  u.h[0] = *(const v8bf*)p0;
  u.h[1] = *(const v8bf*)p1;
  return u.v;
}

// A-matrix 16x32 bf16 K-index pattern (ISA 7.12.2): two contiguous 8-el runs
static __device__ __forceinline__ int a_kidx(int e, int half) {
  int p = e >> 1, c = e & 1;
  return ((p & 3) << 1) + c + ((p >> 2) << 4) + (half << 3);
}

// A fragment from row-major [16 x >=32] bf16; rows = lane&15
static __device__ __forceinline__ v16bf load_a_rm(const __bf16* base, int ld, int lane) {
  int m = lane & 15, half = lane >> 4;
  const __bf16* r = base + m * ld + (half << 3);
  return load2x8(r, r + 16);
}

// B fragment where B[kk][n] = src[n][kk]: two 8-el runs per lane (k = 8h.., 16+8h..)
static __device__ __forceinline__ v16bf load_bT_rows(const __bf16* base, int ld, int lane) {
  int n = lane & 15, half = lane >> 4;
  const __bf16* r = base + n * ld + (half << 3);
  return load2x8(r, r + 16);
}

// ---- conversion / packing kernels -----------------------------------------

__global__ void cvt_f32_bf16(const float* __restrict__ src, __bf16* __restrict__ dst, int n) {
  int i = blockIdx.x * blockDim.x + threadIdx.x;
  if (i < n) dst[i] = (__bf16)src[i];
}

// packs W_{q,k,v} [768 x 768] into transposed [2304 n][768 k] bf16
__global__ void pack_qkv_wT(const float* __restrict__ Wq, const float* __restrict__ Wk,
                            const float* __restrict__ Wv, __bf16* __restrict__ dst) {
  int i = blockIdx.x * blockDim.x + threadIdx.x;   // over 2304*768 outputs
  if (i >= 3 * Dd * Dd) return;
  int n = i / Dd, k = i % Dd;
  int sel = n / Dd, nw = n % Dd;
  const float* W = sel == 0 ? Wq : (sel == 1 ? Wk : Wv);
  dst[i] = (__bf16)W[k * Dd + nw];
}

// Wo [768 k][768 n] -> transposed bf16 [768 n][768 k]
__global__ void cvt_wT(const float* __restrict__ src, __bf16* __restrict__ dst) {
  int i = blockIdx.x * blockDim.x + threadIdx.x;   // over 768*768
  if (i >= Dd * Dd) return;
  int n = i / Dd, k = i % Dd;
  dst[i] = (__bf16)src[k * Dd + n];
}

// rel_emb [129 x 64] -> relb [160 r][64 d] (zero padded) and relbT [64 d][160 r]
__global__ void pad_rel(const float* __restrict__ rel, __bf16* __restrict__ relb,
                        __bf16* __restrict__ relbT) {
  int i = blockIdx.x * blockDim.x + threadIdx.x;   // over 160*64
  if (i >= R2P * DK) return;
  int r = i >> 6, d = i & 63;
  __bf16 v = (r < R2) ? (__bf16)rel[i] : (__bf16)0.0f;
  relb[i] = v;
  relbT[d * R2P + r] = v;
}

// ---- fused QKV projection GEMM: [4096x768] x [768x2304]^T-packed ----------
// q,k -> [B][H][S][64];  v -> transposed [B][H][64][S]

__global__ __launch_bounds__(128) void gemm_qkv(
    const __bf16* __restrict__ A, const __bf16* __restrict__ BwT,
    const float* __restrict__ bq, const float* __restrict__ bk,
    const float* __restrict__ bv,
    __bf16* __restrict__ qh, __bf16* __restrict__ kh, __bf16* __restrict__ vht) {
  __shared__ __align__(16) __bf16 As[2][64 * 40];
  __shared__ __align__(16) __bf16 Bts[2][64 * 40];
  int tid = threadIdx.x, lane = tid & 31, w = tid >> 5;
  int m0 = blockIdx.x * 64, n0 = blockIdx.y * 64;
  const __bf16* Ag = A + m0 * Dd;
  const __bf16* Bg = BwT + n0 * Dd;
  v8f acc[4] = {};
  stage_tile32(Ag, Dd, As[0], tid);
  stage_tile32(Bg, Dd, Bts[0], tid);
  for (int step = 0; step < 24; ++step) {
    stage_wait<0>();
    __syncthreads();
    if (step + 1 < 24) {
      stage_tile32(Ag + (step + 1) * 32, Dd, As[(step + 1) & 1], tid);
      stage_tile32(Bg + (step + 1) * 32, Dd, Bts[(step + 1) & 1], tid);
    }
    const __bf16* as = As[step & 1];
    const __bf16* bs = Bts[step & 1];
    v16bf a  = load_a_rm(as + (w * 16) * 40, 40, lane);
    v16bf b0 = load_bT_rows(bs,            40, lane);
    v16bf b1 = load_bT_rows(bs + 16 * 40,  40, lane);
    v16bf b2 = load_bT_rows(bs + 32 * 40,  40, lane);
    v16bf b3 = load_bT_rows(bs + 48 * 40,  40, lane);
    acc[0] = wmma_bf16(a, b0, acc[0]);
    acc[1] = wmma_bf16(a, b1, acc[1]);
    acc[2] = wmma_bf16(a, b2, acc[2]);
    acc[3] = wmma_bf16(a, b3, acc[3]);
  }
  int n15 = lane & 15, half = lane >> 4;
#pragma unroll
  for (int nt = 0; nt < 4; ++nt) {
    int ng = n0 + nt * 16 + n15;
    int sel = ng / Dd, within = ng % Dd;
    int h = within >> 6, dk = within & 63;
    float bias = (sel == 0 ? bq : (sel == 1 ? bk : bv))[within];
#pragma unroll
    for (int i = 0; i < 8; ++i) {
      int m = m0 + w * 16 + i + 8 * half;
      int b = m >> 10, s = m & 1023;
      int bh = b * Hh + h;
      __bf16 val = (__bf16)(acc[nt][i] + bias);
      if (sel == 0)      qh[(bh * Ss + s) * DK + dk] = val;
      else if (sel == 1) kh[(bh * Ss + s) * DK + dk] = val;
      else               vht[(bh * DK + dk) * Ss + s] = val;   // transposed V
    }
  }
}

// ---- attention: two-pass flash with relative-position bias ----------------
// grid = B*H*(S/32), block = 64 (2 waves x 16 query rows)

__global__ __launch_bounds__(64) void attn_kernel(
    const __bf16* __restrict__ qh, const __bf16* __restrict__ kh,
    const __bf16* __restrict__ vht, const __bf16* __restrict__ relb,
    const __bf16* __restrict__ relbT, const float* __restrict__ vmask,
    __bf16* __restrict__ ctx) {
  __shared__ __align__(16) __bf16 Ks[2][64 * 64];    // [key][dk]
  __shared__ __align__(16) __bf16 Vts[2][64 * 64];   // [dk][key]
  __shared__ __bf16 relDot[2][16][144];
  __shared__ float pRel[2][16][132];
  __shared__ __align__(16) __bf16 Pbuf[2][16 * 64];

  int tid = threadIdx.x, lane = tid & 31, w = tid >> 5;
  int jt = blockIdx.x & 31;
  int bh = blockIdx.x >> 5;
  int b = bh / Hh;
  int h = bh % Hh;
  int j0 = jt * 32 + w * 16;
  int n15 = lane & 15, half = lane >> 4;

  const __bf16* qbase  = qh  + bh * Ss * DK;
  const __bf16* kbase  = kh  + bh * Ss * DK;
  const __bf16* vtbase = vht + bh * DK * Ss;

  int cmax = (jt * 32 + 31) >> 6;

  // prologue: start DMA of first K tile while we compute rel_dot
  stage_tile64(kbase, DK, Ks[0], tid);

  v16bf aq0 = load_a_rm(qbase + j0 * DK, DK, lane);
  v16bf aq1 = load_a_rm(qbase + j0 * DK + 32, DK, lane);

  // rel_dot = q @ rel_emb^T  (16 x 144; cols >=129 are zero via padding)
#pragma unroll
  for (int rt = 0; rt < 9; ++rt) {
    v16bf br0 = load_bT_rows(relb + rt * 16 * DK, DK, lane);
    v16bf br1 = load_bT_rows(relb + rt * 16 * DK + 32, DK, lane);
    v8f d = {};
    d = wmma_bf16(aq0, br0, d);
    d = wmma_bf16(aq1, br1, d);
#pragma unroll
    for (int i = 0; i < 8; ++i)
      relDot[w][i + 8 * half][rt * 16 + n15] = (__bf16)d[i];
  }
  for (int idx = lane; idx < 16 * 132; idx += 32) (&pRel[w][0][0])[idx] = 0.0f;

  float m_i[8], l_i[8];
#pragma unroll
  for (int i = 0; i < 8; ++i) { m_i[i] = -3.0e38f; l_i[i] = 0.0f; }

  // -------- pass 1: exact row max + softmax denominator --------
  for (int c = 0; c <= cmax; ++c) {
    stage_wait<0>();
    __syncthreads();
    if (c + 1 <= cmax)
      stage_tile64(kbase + (c + 1) * 64 * DK, DK, Ks[(c + 1) & 1], tid);
    const __bf16* ks = Ks[c & 1];
    int kb = c * 64;
    float sc[32];
#pragma unroll
    for (int nt = 0; nt < 4; ++nt) {
      v16bf bk0 = load_bT_rows(ks + nt * 16 * 64, 64, lane);
      v16bf bk1 = load_bT_rows(ks + nt * 16 * 64 + 32, 64, lane);
      v8f d = {};
      d = wmma_bf16(aq0, bk0, d);
      d = wmma_bf16(aq1, bk1, d);
      int key = kb + nt * 16 + n15;
      float km = (1.0f - vmask[b * Ss + key]) * NEGC;
#pragma unroll
      for (int i = 0; i < 8; ++i) {
        int j = j0 + i + 8 * half;
        int dji = key - j; dji = dji < -64 ? -64 : (dji > 64 ? 64 : dji);
        float s = (d[i] + (float)relDot[w][i + 8 * half][dji + 64]) * 0.125f;
        if (key > j) s -= NEGC;
        s -= km;
        sc[nt * 8 + i] = s;
      }
    }
#pragma unroll
    for (int i = 0; i < 8; ++i) {
      float mx = fmaxf(fmaxf(sc[i], sc[8 + i]), fmaxf(sc[16 + i], sc[24 + i]));
      mx = fmaxf(mx, __shfl_xor(mx, 1, 32));
      mx = fmaxf(mx, __shfl_xor(mx, 2, 32));
      mx = fmaxf(mx, __shfl_xor(mx, 4, 32));
      mx = fmaxf(mx, __shfl_xor(mx, 8, 32));
      float mnew = fmaxf(m_i[i], mx);
      float ss = exp2f((sc[i] - mnew) * L2E) + exp2f((sc[8 + i] - mnew) * L2E) +
                 exp2f((sc[16 + i] - mnew) * L2E) + exp2f((sc[24 + i] - mnew) * L2E);
      ss += __shfl_xor(ss, 1, 32);
      ss += __shfl_xor(ss, 2, 32);
      ss += __shfl_xor(ss, 4, 32);
      ss += __shfl_xor(ss, 8, 32);
      l_i[i] = l_i[i] * exp2f((m_i[i] - mnew) * L2E) + ss;
      m_i[i] = mnew;
    }
  }
  float inv_l[8];
#pragma unroll
  for (int i = 0; i < 8; ++i) inv_l[i] = 1.0f / l_i[i];

  v8f oacc[4] = {};

  // -------- pass 2: normalized p, o += p@v, p_rel scatter --------
  __syncthreads();                       // pass-1 reads done before re-staging
  stage_tile64(kbase, DK, Ks[0], tid);
  stage_tile64(vtbase, Ss, Vts[0], tid);
  for (int c = 0; c <= cmax; ++c) {
    stage_wait<0>();
    __syncthreads();
    if (c + 1 <= cmax) {
      stage_tile64(kbase + (c + 1) * 64 * DK, DK, Ks[(c + 1) & 1], tid);
      stage_tile64(vtbase + (c + 1) * 64, Ss, Vts[(c + 1) & 1], tid);
    }
    const __bf16* ks = Ks[c & 1];
    const __bf16* vs = Vts[c & 1];
    int kb = c * 64;
#pragma unroll
    for (int nt = 0; nt < 4; ++nt) {
      v16bf bk0 = load_bT_rows(ks + nt * 16 * 64, 64, lane);
      v16bf bk1 = load_bT_rows(ks + nt * 16 * 64 + 32, 64, lane);
      v8f d = {};
      d = wmma_bf16(aq0, bk0, d);
      d = wmma_bf16(aq1, bk1, d);
      int key = kb + nt * 16 + n15;
      float km = (1.0f - vmask[b * Ss + key]) * NEGC;
#pragma unroll
      for (int i = 0; i < 8; ++i) {
        int j = j0 + i + 8 * half;
        int dji = key - j; dji = dji < -64 ? -64 : (dji > 64 ? 64 : dji);
        int pos = dji + 64;
        float s = (d[i] + (float)relDot[w][i + 8 * half][pos]) * 0.125f;
        if (key > j) s -= NEGC;
        s -= km;
        float pv = exp2f((s - m_i[i]) * L2E) * inv_l[i];
        atomicAdd(&pRel[w][i + 8 * half][pos], pv);
        Pbuf[w][(i + 8 * half) * 64 + nt * 16 + n15] = (__bf16)pv;
      }
    }
    // Pbuf[w] is per-wave; same-wave LDS ops stay in order -> no barrier
    v16bf ap0 = load_a_rm(&Pbuf[w][0], 64, lane);
    v16bf ap1 = load_a_rm(&Pbuf[w][32], 64, lane);
#pragma unroll
    for (int no = 0; no < 4; ++no) {
      v16bf bv0 = load_bT_rows(vs + (no * 16) * 64, 64, lane);
      v16bf bv1 = load_bT_rows(vs + (no * 16) * 64 + 32, 64, lane);
      oacc[no] = wmma_bf16(ap0, bv0, oacc[no]);
      oacc[no] = wmma_bf16(ap1, bv1, oacc[no]);
    }
  }
  __syncthreads();

  // -------- o += p_rel @ rel_emb  (K = 129, 5 x 32 steps, zero-padded) -----
#pragma unroll
  for (int t = 0; t < 5; ++t) {
    v16bf ar;
#pragma unroll
    for (int e = 0; e < 16; ++e) {
      int r = t * 32 + a_kidx(e, half);
      ar[e] = (__bf16)((r < R2) ? pRel[w][n15][r] : 0.0f);
    }
#pragma unroll
    for (int no = 0; no < 4; ++no)
      oacc[no] = wmma_bf16(ar, load_bT_rows(relbT + (no * 16) * R2P + t * 32, R2P, lane),
                           oacc[no]);
  }

  // -------- write context [B*S][768] bf16 ----------------------------------
#pragma unroll
  for (int no = 0; no < 4; ++no) {
#pragma unroll
    for (int i = 0; i < 8; ++i) {
      int j = j0 + i + 8 * half;
      ctx[(b * Ss + j) * Dd + h * DK + no * 16 + n15] = (__bf16)oacc[no][i];
    }
  }
}

// ---- output projection GEMM: ctx[4096x768] x WoT[768n][768k] + bo, mask ---

__global__ __launch_bounds__(128) void gemm_out(
    const __bf16* __restrict__ A, const __bf16* __restrict__ BwT,
    const float* __restrict__ bo, const float* __restrict__ vmask,
    float* __restrict__ out) {
  __shared__ __align__(16) __bf16 As[2][64 * 40];
  __shared__ __align__(16) __bf16 Bts[2][64 * 40];
  int tid = threadIdx.x, lane = tid & 31, w = tid >> 5;
  int m0 = blockIdx.x * 64, n0 = blockIdx.y * 64;
  const __bf16* Ag = A + m0 * Dd;
  const __bf16* Bg = BwT + n0 * Dd;
  v8f acc[4] = {};
  stage_tile32(Ag, Dd, As[0], tid);
  stage_tile32(Bg, Dd, Bts[0], tid);
  for (int step = 0; step < 24; ++step) {
    stage_wait<0>();
    __syncthreads();
    if (step + 1 < 24) {
      stage_tile32(Ag + (step + 1) * 32, Dd, As[(step + 1) & 1], tid);
      stage_tile32(Bg + (step + 1) * 32, Dd, Bts[(step + 1) & 1], tid);
    }
    const __bf16* as = As[step & 1];
    const __bf16* bs = Bts[step & 1];
    v16bf a  = load_a_rm(as + (w * 16) * 40, 40, lane);
    v16bf b0 = load_bT_rows(bs,            40, lane);
    v16bf b1 = load_bT_rows(bs + 16 * 40,  40, lane);
    v16bf b2 = load_bT_rows(bs + 32 * 40,  40, lane);
    v16bf b3 = load_bT_rows(bs + 48 * 40,  40, lane);
    acc[0] = wmma_bf16(a, b0, acc[0]);
    acc[1] = wmma_bf16(a, b1, acc[1]);
    acc[2] = wmma_bf16(a, b2, acc[2]);
    acc[3] = wmma_bf16(a, b3, acc[3]);
  }
  int n15 = lane & 15, half = lane >> 4;
#pragma unroll
  for (int nt = 0; nt < 4; ++nt) {
    int ng = n0 + nt * 16 + n15;
    float bias = bo[ng];
#pragma unroll
    for (int i = 0; i < 8; ++i) {
      int m = m0 + w * 16 + i + 8 * half;
      out[m * Dd + ng] = (acc[nt][i] + bias) * vmask[m];
    }
  }
}

// ---------------------------------------------------------------------------

extern "C" void kernel_launch(void* const* d_in, const int* in_sizes, int n_in,
                              void* d_out, int out_size, void* d_ws, size_t ws_size,
                              hipStream_t stream) {
  (void)in_sizes; (void)n_in; (void)out_size; (void)ws_size;
  const float* x     = (const float*)d_in[0];
  const float* vmask = (const float*)d_in[1];
  const float* rel   = (const float*)d_in[2];
  const float* Wq    = (const float*)d_in[3];
  const float* bq    = (const float*)d_in[4];
  const float* Wk    = (const float*)d_in[5];
  const float* bk    = (const float*)d_in[6];
  const float* Wv    = (const float*)d_in[7];
  const float* bv    = (const float*)d_in[8];
  const float* Wo    = (const float*)d_in[9];
  const float* bo    = (const float*)d_in[10];
  float* out = (float*)d_out;

  char* p = (char*)d_ws;
  __bf16* xb    = (__bf16*)p; p += (size_t)Bb * Ss * Dd * 2;          // 4096x768
  __bf16* wqkvT = (__bf16*)p; p += (size_t)3 * Dd * Dd * 2;           // [2304 n][768 k]
  __bf16* wobT  = (__bf16*)p; p += (size_t)Dd * Dd * 2;               // [768 n][768 k]
  __bf16* relb  = (__bf16*)p; p += (size_t)R2P * DK * 2;              // [160 r][64 d]
  __bf16* relbT = (__bf16*)p; p += (size_t)DK * R2P * 2;              // [64 d][160 r]
  __bf16* qh    = (__bf16*)p; p += (size_t)Bb * Hh * Ss * DK * 2;     // [B][H][S][64]
  __bf16* kh    = (__bf16*)p; p += (size_t)Bb * Hh * Ss * DK * 2;     // [B][H][S][64]
  __bf16* vht   = (__bf16*)p; p += (size_t)Bb * Hh * DK * Ss * 2;     // [B][H][64][S]
  __bf16* ctx   = (__bf16*)p; p += (size_t)Bb * Ss * Dd * 2;          // 4096x768

  int nx = Bb * Ss * Dd;
  cvt_f32_bf16<<<(nx + 255) / 256, 256, 0, stream>>>(x, xb, nx);
  int nw = 3 * Dd * Dd;
  pack_qkv_wT<<<(nw + 255) / 256, 256, 0, stream>>>(Wq, Wk, Wv, wqkvT);
  int no = Dd * Dd;
  cvt_wT<<<(no + 255) / 256, 256, 0, stream>>>(Wo, wobT);
  pad_rel<<<(R2P * DK + 255) / 256, 256, 0, stream>>>(rel, relb, relbT);

  gemm_qkv<<<dim3(Bb * Ss / 64, 3 * Dd / 64), 128, 0, stream>>>(
      xb, wqkvT, bq, bk, bv, qh, kh, vht);

  attn_kernel<<<Bb * Hh * (Ss / 32), 64, 0, stream>>>(
      qh, kh, vht, relb, relbT, vmask, ctx);

  gemm_out<<<dim3(Bb * Ss / 64, Dd / 64), 128, 0, stream>>>(
      ctx, wobT, bo, vmask, out);
}
